// LatentGNN_Kernel_ch_69544110456901
// MI455X (gfx1250) — matrix operand order repaired and checked by reading, not compiled, not run on hardware
//
#include <hip/hip_runtime.h>
#include <hip/hip_bf16.h>

typedef __bf16 bf16;
typedef __attribute__((ext_vector_type(16))) __bf16 v16bf;
typedef __attribute__((ext_vector_type(8)))  __bf16 v8bf;
typedef __attribute__((ext_vector_type(2)))  __bf16 v2bf;
typedef __attribute__((ext_vector_type(8)))  float  v8f;
typedef __attribute__((ext_vector_type(4)))  float  v4f;

// Problem dims (fixed by the reference)
static constexpr int Bn  = 16;
static constexpr int Cc  = 2048;
static constexpr int HWn = 1024;
static constexpr int Dn  = 256;

// ---------------------------------------------------------------------------
// gfx1250 async copy: per-lane global -> LDS, tracked by ASYNCcnt.
// VDST operand is the LDS byte offset; for generic pointers to __shared__
// data the low 32 bits are the LDS offset (shared aperture lives in the
// high word -- see CDNA5 ISA aperture mapping).
// ---------------------------------------------------------------------------
__device__ __forceinline__ void async_copy_b128(const void* gptr, void* lptr) {
    unsigned loff = (unsigned)(uintptr_t)lptr;
    asm volatile("global_load_async_to_lds_b128 %0, %1, off"
                 :: "v"(loff), "v"(gptr)
                 : "memory");
}
__device__ __forceinline__ void wait_async0() {
    asm volatile("s_wait_asynccnt 0x0" ::: "memory");
}

// ---------------------------------------------------------------------------
// Fragment loaders.  NT GEMM: C[M,N] = A[M,K] * Bt[N,K]^T, K-contiguous both.
// A 16x32 bf16 fragment per ISA table: lane L -> row (L&15),
//   K = (L>>4)*8 + {0..7} and + {16..23}   (two 16-byte chunks)
// B 32x16 fragment (B^T rows): lane L -> col (L&15),
//   K = (L>>4)*8 + {0..7} and + {16..23}   (same as A)
// ---------------------------------------------------------------------------
__device__ __forceinline__ v16bf loadA(const bf16* __restrict__ row, int k0, int lane) {
    const bf16* p = row + k0 + ((lane >> 4) * 8);
    v8bf lo = *(const v8bf*)(p);
    v8bf hi = *(const v8bf*)(p + 16);
    return __builtin_shufflevector(lo, hi, 0,1,2,3,4,5,6,7,8,9,10,11,12,13,14,15);
}
__device__ __forceinline__ v16bf loadA(const float* __restrict__ row, int k0, int lane) {
    const float* p = row + k0 + ((lane >> 4) * 8);
    v4f f0 = *(const v4f*)(p);
    v4f f1 = *(const v4f*)(p + 4);
    v4f f2 = *(const v4f*)(p + 16);
    v4f f3 = *(const v4f*)(p + 20);
    v16bf r;
#pragma unroll
    for (int i = 0; i < 4; ++i) {
        r[i]      = (bf16)f0[i];
        r[4 + i]  = (bf16)f1[i];
        r[8 + i]  = (bf16)f2[i];
        r[12 + i] = (bf16)f3[i];
    }
    return r;
}
__device__ __forceinline__ v16bf loadB(const float* __restrict__ row, int k0, int lane) {
    const float* p = row + k0 + ((lane >> 4) * 8);
    v16bf r;
#pragma unroll
    for (int i = 0; i < 16; i += 4) {
        v4f f = *(const v4f*)(p + i + (i & 8));
        r[i] = (bf16)f[0]; r[i+1] = (bf16)f[1]; r[i+2] = (bf16)f[2]; r[i+3] = (bf16)f[3];
    }
    return r;
}

// Common epilogue: C/D layout -> element i of v8f is row (i + (lane>=16?8:0)),
// col (lane&15).  EPI: 0 = f32 store, 1 = BN+ReLU by row, 2 = BN+ReLU by col,
// 3 = f32 + transposed bf16, 4 = transposed bf16 only.
template <int EPI>
__device__ __forceinline__ void epilogue(
    const v8f (&acc)[4], int b, int m0, int n0, int lane,
    float* __restrict__ O, long sOb, int ldo,
    bf16* __restrict__ OT, long sOTb, int ldot,
    const float* __restrict__ gamma, const float* __restrict__ beta)
{
    const int mlo = (lane >> 4) * 8;
    const int nl  = lane & 15;
#pragma unroll
    for (int t = 0; t < 4; ++t) {
#pragma unroll
        for (int i = 0; i < 8; ++i) {
            int m = m0 + mlo + i;
            int n = n0 + t * 16 + nl;
            float v = acc[t][i];
            if constexpr (EPI == 1) {
                float s = gamma[m] * rsqrtf(1.0f + 1e-5f);
                v = fmaxf(v * s + beta[m], 0.0f);
            }
            if constexpr (EPI == 2) {
                float s = gamma[n] * rsqrtf(1.0f + 1e-5f);
                v = fmaxf(v * s + beta[n], 0.0f);
            }
            if constexpr (EPI == 0 || EPI == 1 || EPI == 2 || EPI == 3)
                O[(long)b * sOb + (long)m * ldo + n] = v;
            if constexpr (EPI == 3 || EPI == 4)
                OT[(long)b * sOTb + (long)n * ldot + m] = (bf16)v;
        }
    }
}

// ---------------------------------------------------------------------------
// Direct NT GEMM for f32 sources (psi projections): global_load_b128 ->
// v_cvt_pk_bf16_f32 -> WMMA.  Block = 8 wave32s (4M x 2N), 16x64 per wave.
// ---------------------------------------------------------------------------
template <int EPI>
__global__ __launch_bounds__(256) void gemm_nt_f32(
    const float* __restrict__ A, long sAb, int lda,
    const float* __restrict__ Bt, long sBb, int ldb,
    float* __restrict__ O, long sOb, int ldo,
    int K,
    const float* __restrict__ gamma, const float* __restrict__ beta)
{
    const int lane = threadIdx.x & 31;
    const int wave = threadIdx.x >> 5;
    const int wm = wave >> 1;
    const int wn = wave & 1;
    const int b  = blockIdx.z;
    const int m0 = blockIdx.y * 64 + wm * 16;
    const int n0 = blockIdx.x * 128 + wn * 64;
    const int r  = lane & 15;

    A  += (long)b * sAb;
    Bt += (long)b * sBb;

    v8f acc[4];
#pragma unroll
    for (int t = 0; t < 4; ++t)
#pragma unroll
        for (int i = 0; i < 8; ++i) acc[t][i] = 0.0f;

    const float* Arow  = A  + (long)(m0 + r) * lda;
    const float* Brow0 = Bt + (long)(n0 + r) * ldb;

    for (int k = 0; k < K; k += 32) {
        v16bf afrag = loadA(Arow, k, lane);
#pragma unroll
        for (int t = 0; t < 4; ++t) {
            v16bf bfrag = loadB(Brow0 + (long)(t * 16) * ldb, k, lane);
            acc[t] = __builtin_amdgcn_wmma_f32_16x16x32_bf16(
                false, afrag, false, bfrag, (short)0, acc[t], false, false);
        }
    }
    epilogue<EPI>(acc, b, m0, n0, lane, O, sOb, ldo, nullptr, 0, 0, gamma, beta);
}

// ---------------------------------------------------------------------------
// LDS-staged NT GEMM for bf16 sources.  Per K-step the block stages a 64x32 A
// tile (4 KB) and a 128x32 B^T tile (8 KB) into LDS with per-lane
// global_load_async_to_lds_b128 (ASYNCcnt), then every wave feeds its WMMA
// fragments from LDS (ds_load_b128).  Global B traffic drops 4x vs direct.
// ---------------------------------------------------------------------------
template <int EPI>
__global__ __launch_bounds__(256) void gemm_nt_lds(
    const bf16* __restrict__ A, long sAb, int lda,
    const bf16* __restrict__ Bt, long sBb, int ldb,
    float* __restrict__ O, long sOb, int ldo,
    bf16* __restrict__ OT, long sOTb, int ldot,
    int K)
{
    __shared__ bf16 sA[64 * 32];    // ld = 32 halves (64 B rows)
    __shared__ bf16 sB[128 * 32];   // ld = 32 halves

    const int tid  = threadIdx.x;
    const int lane = tid & 31;
    const int wave = tid >> 5;
    const int wm = wave >> 1;
    const int wn = wave & 1;
    const int b  = blockIdx.z;
    const int mblk = blockIdx.y * 64;
    const int nblk = blockIdx.x * 128;
    const int r  = lane & 15;

    A  += (long)b * sAb;
    Bt += (long)b * sBb;

    // staging coordinates (16-byte chunks of 8 halves)
    const int arow = tid >> 2, ach = tid & 3;              // 64 rows x 4 chunks
    const int brow = tid >> 1, bch = (tid & 1) * 2;        // 128 rows x 4 chunks, 2/thread
    const bf16* gA = A  + (long)(mblk + arow) * lda + ach * 8;
    const bf16* gB = Bt + (long)(nblk + brow) * ldb + bch * 8;
    bf16* lA  = &sA[arow * 32 + ach * 8];
    bf16* lB0 = &sB[brow * 32 + bch * 8];

    v8f acc[4];
#pragma unroll
    for (int t = 0; t < 4; ++t)
#pragma unroll
        for (int i = 0; i < 8; ++i) acc[t][i] = 0.0f;

    const bf16* fA  = &sA[(wm * 16 + r) * 32];             // A fragment row
    const bf16* fB0 = &sB[(wn * 64 + r) * 32];             // B fragment row base

    for (int k = 0; k < K; k += 32) {
        __syncthreads();                        // previous-tile reads done
        async_copy_b128(gA + k,      lA);
        async_copy_b128(gB + k,      lB0);
        async_copy_b128(gB + k + 16, lB0 + 16);
        if (k + 32 < K) {                       // warm GL2 for next K-step
            __builtin_prefetch(gA + k + 32, 0, 1);
            __builtin_prefetch(gB + k + 32, 0, 1);
        }
        wait_async0();                          // this wave's copies landed
        __syncthreads();                        // all waves' copies landed

        v16bf afrag = loadA(fA, 0, lane);
#pragma unroll
        for (int t = 0; t < 4; ++t) {
            const bf16* pb = fB0 + t * 16 * 32 + (lane >> 4) * 8;
            v8bf blo = *(const v8bf*)(pb);
            v8bf bhi = *(const v8bf*)(pb + 16);
            v16bf bfrag = __builtin_shufflevector(blo, bhi, 0,1,2,3,4,5,6,7,8,9,10,11,12,13,14,15);
            acc[t] = __builtin_amdgcn_wmma_f32_16x16x32_bf16(
                false, afrag, false, bfrag, (short)0, acc[t], false, false);
        }
    }
    epilogue<EPI>(acc, b, mblk + wm * 16, nblk + wn * 64, lane,
                  O, sOb, ldo, OT, sOTb, ldot, nullptr, nullptr);
}

// ---------------------------------------------------------------------------
// Mt[b][h][c] = input[b][(c&1)*1024 + h][c>>1]  (the torch raw-reinterpret
// view, materialized directly in the B^T layout the latent GEMM wants).
// ---------------------------------------------------------------------------
__global__ __launch_bounds__(256) void build_mt(const float* __restrict__ in,
                                                bf16* __restrict__ mt)
{
    long t  = (long)blockIdx.x * 256 + threadIdx.x;   // [0, 16*1024*1024)
    int sp  = (int)(t & 1023);
    int h   = (int)((t >> 10) & 1023);
    int b   = (int)(t >> 20);
    float a0 = in[((long)b * Cc + h) * HWn + sp];
    float a1 = in[((long)b * Cc + HWn + h) * HWn + sp];
    v2bf pk; pk[0] = (bf16)a0; pk[1] = (bf16)a1;
    *(v2bf*)(mt + ((long)b * HWn + h) * Cc + 2 * sp) = pk;
}

// ---------------------------------------------------------------------------
// Per-row L2 normalization, f32 in -> bf16 out.  One 256-thread block per row.
// ---------------------------------------------------------------------------
__global__ __launch_bounds__(256) void rownorm_bf16(const float* __restrict__ X,
                                                    bf16* __restrict__ Y, int L)
{
    __shared__ float red[8];
    long base = (long)blockIdx.x * L;
    int tid = threadIdx.x;
    float ss = 0.0f;
    for (int i = tid; i < L; i += 256) { float v = X[base + i]; ss += v * v; }
    for (int off = 16; off > 0; off >>= 1) ss += __shfl_down(ss, off, 32);
    if ((tid & 31) == 0) red[tid >> 5] = ss;
    __syncthreads();
    float tot = 0.0f;
#pragma unroll
    for (int i = 0; i < 8; ++i) tot += red[i];
    float inv = 1.0f / fmaxf(sqrtf(tot), 1e-12f);
    for (int i = tid; i < L; i += 256) Y[base + i] = (bf16)(X[base + i] * inv);
}

// ---------------------------------------------------------------------------
// Row softmax over length-256 rows, f32 in -> bf16 out. One element/thread.
// ---------------------------------------------------------------------------
__global__ __launch_bounds__(256) void softmax_bf16(const float* __restrict__ X,
                                                    bf16* __restrict__ Y)
{
    __shared__ float red[8];
    long base = (long)blockIdx.x * 256;
    int tid = threadIdx.x;
    float v = X[base + tid];
    float m = v;
    for (int off = 16; off > 0; off >>= 1) m = fmaxf(m, __shfl_xor(m, off, 32));
    if ((tid & 31) == 0) red[tid >> 5] = m;
    __syncthreads();
    float gm = red[0];
#pragma unroll
    for (int i = 1; i < 8; ++i) gm = fmaxf(gm, red[i]);
    float e = __expf(v - gm);
    __syncthreads();
    float s = e;
    for (int off = 16; off > 0; off >>= 1) s += __shfl_xor(s, off, 32);
    if ((tid & 31) == 0) red[tid >> 5] = s;
    __syncthreads();
    float tot = 0.0f;
#pragma unroll
    for (int i = 0; i < 8; ++i) tot += red[i];
    Y[base + tid] = (bf16)(e / tot);
}

// ---------------------------------------------------------------------------
extern "C" void kernel_launch(void* const* d_in, const int* in_sizes, int n_in,
                              void* d_out, int out_size, void* d_ws, size_t ws_size,
                              hipStream_t stream)
{
    const float* in_v  = (const float*)d_in[0];
    const float* in_l  = (const float*)d_in[1];
    const float* w_v2l = (const float*)d_in[2];
    const float* g_v2l = (const float*)d_in[3];
    const float* b_v2l = (const float*)d_in[4];
    const float* w_l2v = (const float*)d_in[5];
    const float* g_l2v = (const float*)d_in[6];
    const float* b_l2v = (const float*)d_in[7];
    float* out = (float*)d_out;

    char* ws = (char*)d_ws;
    size_t off = 0;
    auto alloc = [&](size_t bytes) -> void* {
        void* p = ws + off;
        off += (bytes + 255) & ~(size_t)255;
        return p;
    };
    float* Ypsi_v = (float*)alloc((size_t)Bn * Dn * Cc * 4);   // [b][d][c] BN+ReLU'd
    float* Ypsi_l = (float*)alloc((size_t)Bn * Cc * Dn * 4);   // [b][c][d] BN+ReLU'd
    bf16*  Mt     = (bf16*) alloc((size_t)Bn * HWn * Cc * 2);  // B^T for latent GEMM
    bf16*  adj_v  = (bf16*) alloc((size_t)Bn * Dn * Cc * 2);   // [b][d][c]
    bf16*  adj_lT = (bf16*) alloc((size_t)Bn * Cc * Dn * 2);   // [b][c][d]
    float* lat    = (float*)alloc((size_t)Bn * Dn * HWn * 4);  // latent f32 [b][d][h]
    bf16*  latT   = (bf16*) alloc((size_t)Bn * HWn * Dn * 2);  // latent^T bf16 [b][h][d]
    bf16*  latn   = (bf16*) alloc((size_t)Bn * Dn * HWn * 2);  // latent_n [b][d][h]
    float* gram   = (float*)alloc((size_t)Bn * Dn * Dn * 4);   // [b][d][e]
    bf16*  aff    = (bf16*) alloc((size_t)Bn * Dn * Dn * 2);   // softmax(gram)
    bf16*  lat2T  = (bf16*) alloc((size_t)Bn * HWn * Dn * 2);  // (aff*latent)^T

    const long sIn = (long)Cc * HWn;   // 2097152
    const long sDC = (long)Dn * Cc;    // 524288
    const long sDH = (long)Dn * HWn;   // 262144
    const long sDD = (long)Dn * Dn;    // 65536

    // 1) Materialize the raw-reinterpret transpose buffer (bf16)
    build_mt<<<dim3((Bn * HWn * HWn) / 256), 256, 0, stream>>>(in_v, Mt);

    // 2) psi v2l: [D x C] = W_v2l[D,HW] x in_v[C,HW]^T, BN+ReLU by row (d)
    gemm_nt_f32<1><<<dim3(Cc / 128, Dn / 64, Bn), 256, 0, stream>>>(
        w_v2l, 0, HWn, in_v, sIn, HWn, Ypsi_v, sDC, Cc, HWn, g_v2l, b_v2l);

    // 3) psi l2v: [C x D] = in_l[C,HW] x W_l2v[D,HW]^T, BN+ReLU by col (d)
    gemm_nt_f32<2><<<dim3(Dn / 128, Cc / 64, Bn), 256, 0, stream>>>(
        in_l, sIn, HWn, w_l2v, 0, HWn, Ypsi_l, sDC, Dn, HWn, g_l2v, b_l2v);

    // 4) v2l_adj: normalize over c (rows of [D][C])
    rownorm_bf16<<<dim3(Bn * Dn), 256, 0, stream>>>(Ypsi_v, adj_v, Cc);
    // 5) l2v_adj^T: normalize over d (rows of [C][D])
    rownorm_bf16<<<dim3(Bn * Cc), 256, 0, stream>>>(Ypsi_l, adj_lT, Dn);

    // 6) latent: [D x HW] = adj_v[D,C] x Mt[HW,C]^T ; dual store f32 + bf16^T
    gemm_nt_lds<3><<<dim3(HWn / 128, Dn / 64, Bn), 256, 0, stream>>>(
        adj_v, sDC, Cc, Mt, sIn, Cc, lat, sDH, HWn, latT, sDH, Dn, Cc);

    // 7) latent_n: normalize rows of [D][HW]
    rownorm_bf16<<<dim3(Bn * Dn), 256, 0, stream>>>(lat, latn, HWn);

    // 8) gram: [D x D] = latn x latn^T
    gemm_nt_lds<0><<<dim3(Dn / 128, Dn / 64, Bn), 256, 0, stream>>>(
        latn, sDH, HWn, latn, sDH, HWn, gram, sDD, Dn, nullptr, 0, 0, HWn);

    // 9) aff = softmax(gram) rows, bf16
    softmax_bf16<<<dim3(Bn * Dn), 256, 0, stream>>>(gram, aff);

    // 10) latent2^T: [HW x D] store of aff[D,D] x latT[HW,D]^T
    gemm_nt_lds<4><<<dim3(HWn / 128, Dn / 64, Bn), 256, 0, stream>>>(
        aff, sDD, Dn, latT, sDH, Dn, nullptr, 0, 0, lat2T, sDH, Dn, Dn);

    // 11) visible: [C x HW] = adj_lT[C,D] x lat2T[HW,D]^T -> d_out
    gemm_nt_lds<0><<<dim3(HWn / 128, Cc / 64, Bn), 256, 0, stream>>>(
        adj_lT, sDC, Dn, lat2T, sDH, Dn, out, sIn, HWn, nullptr, 0, 0, Dn);

    (void)in_sizes; (void)n_in; (void)out_size; (void)ws_size;
}
